// GATModel_40862318854872
// MI455X (gfx1250) — matrix operand order repaired and checked by reading, not compiled, not run on hardware
//
#include <hip/hip_runtime.h>
#include <hip/hip_bf16.h>

typedef __attribute__((ext_vector_type(16))) _Float16 v16h;
typedef __attribute__((ext_vector_type(8)))  float    v8f;

#define NN 50000
#define NE 800000
#define INF_ 256
#define DH 64

// order-preserving float->uint key for atomic max over signed floats
__device__ __forceinline__ unsigned fkey(float x) {
    unsigned u = __float_as_uint(x);
    return (u & 0x80000000u) ? ~u : (u | 0x80000000u);
}
__device__ __forceinline__ float fkeyinv(unsigned k) {
    unsigned u = (k & 0x80000000u) ? (k & 0x7FFFFFFFu) : ~k;
    return __uint_as_float(u);
}

__global__ void zero_f(float* __restrict__ p, size_t n) {
    size_t i = (size_t)blockIdx.x * blockDim.x + threadIdx.x;
    if (i < n) p[i] = 0.0f;
}

// ---------------------------------------------------------------------------
// Projection: h_cat[N,128] = x @ [W_src | W_dst]  (f16 WMMA, f32 accum)
// plus fused a_src = h_src . att_src, a_dst = h_dst . att_dst
// Block: 256 threads = 8 waves; each wave computes a 16-row tile.
// ---------------------------------------------------------------------------
__global__ __launch_bounds__(256) void gat_proj(
    const float* __restrict__ x,
    const float* __restrict__ Wsrc, const float* __restrict__ Wdst,
    const float* __restrict__ att_s, const float* __restrict__ att_d,
    float* __restrict__ hcat, float* __restrict__ a_s, float* __restrict__ a_d)
{
    __shared__ _Float16 WhT[128 * 256];   // WhT[c*256 + k] = Wcat[k][c], 64 KB
    const int tid = threadIdx.x;
    for (int i = tid; i < 128 * 256; i += 256) {
        int c = i & 127, k = i >> 7;      // coalesced read over c
        float w = (c < 64) ? Wsrc[k * 64 + c] : Wdst[k * 64 + (c - 64)];
        WhT[c * 256 + k] = (_Float16)w;
    }
    __syncthreads();

    const int wave = tid >> 5;
    const int lane = tid & 31;
    const int h    = lane >> 4;           // lane half
    const int ln   = lane & 15;
    const int rowbase = blockIdx.x * 128 + wave * 16;

    int arow = rowbase + ln;              // A-matrix row (both halves load same rows)
    if (arow >= NN) arow = NN - 1;        // clamp: keep EXEC all-ones for WMMA
    const float* xr = x + (size_t)arow * INF_;

    v8f acc[8];
    #pragma unroll
    for (int t = 0; t < 8; ++t)
        #pragma unroll
        for (int r = 0; r < 8; ++r) acc[t][r] = 0.0f;

    #pragma unroll
    for (int kt = 0; kt < 8; ++kt) {
        // A fragment, k order chosen here (same as B): element e -> K = kt*32 + 16h + e (contiguous)
        const float* p = xr + kt * 32 + 16 * h;
        float4 f0 = *(const float4*)(p);
        float4 f1 = *(const float4*)(p + 4);
        float4 f2 = *(const float4*)(p + 8);
        float4 f3 = *(const float4*)(p + 12);
        v16h a;
        a[0]=(_Float16)f0.x; a[1]=(_Float16)f0.y; a[2]=(_Float16)f0.z; a[3]=(_Float16)f0.w;
        a[4]=(_Float16)f1.x; a[5]=(_Float16)f1.y; a[6]=(_Float16)f1.z; a[7]=(_Float16)f1.w;
        a[8]=(_Float16)f2.x; a[9]=(_Float16)f2.y; a[10]=(_Float16)f2.z; a[11]=(_Float16)f2.w;
        a[12]=(_Float16)f3.x; a[13]=(_Float16)f3.y; a[14]=(_Float16)f3.z; a[15]=(_Float16)f3.w;

        #pragma unroll
        for (int nt = 0; nt < 8; ++nt) {
            // B fragment: lane column N = nt*16+ln, K = kt*32 + 16h + e (contiguous)
            int n = nt * 16 + ln;
            v16h b = *(const v16h*)(&WhT[n * 256 + kt * 32 + 16 * h]);
            acc[nt] = __builtin_amdgcn_wmma_f32_16x16x32_f16(
                false, a, false, b, (short)0, acc[nt], false, false);
        }
    }

    // fused attention dot products: per lane, partial over its 8 columns
    float ps[8], pd[8];
    #pragma unroll
    for (int r = 0; r < 8; ++r) { ps[r] = 0.0f; pd[r] = 0.0f; }
    #pragma unroll
    for (int nt = 0; nt < 4; ++nt) {
        float as_ = att_s[nt * 16 + ln];
        float ad_ = att_d[nt * 16 + ln];
        #pragma unroll
        for (int r = 0; r < 8; ++r) {
            ps[r] += acc[nt][r]     * as_;
            pd[r] += acc[nt + 4][r] * ad_;
        }
    }
    // reduce across the 16 lanes of each half (same C-matrix row)
    #pragma unroll
    for (int m = 8; m >= 1; m >>= 1) {
        #pragma unroll
        for (int r = 0; r < 8; ++r) {
            ps[r] += __shfl_xor(ps[r], m, 16);
            pd[r] += __shfl_xor(pd[r], m, 16);
        }
    }

    // store: C/D layout row = 8h + r, col = nt*16 + ln
    #pragma unroll
    for (int r = 0; r < 8; ++r) {
        int row = rowbase + 8 * h + r;
        if (row < NN) {
            float* hr = hcat + (size_t)row * 128;
            #pragma unroll
            for (int nt = 0; nt < 8; ++nt) hr[nt * 16 + ln] = acc[nt][r];
            if (ln == 0) { a_s[row] = ps[r]; a_d[row] = pd[r]; }
        }
    }
}

// ---------------------------------------------------------------------------
// Edge kernels
// ---------------------------------------------------------------------------
__global__ void edge_score(const int* __restrict__ ei,
                           const float* __restrict__ a_s, const float* __restrict__ a_d,
                           float* __restrict__ esc, unsigned* __restrict__ key)
{
    int i = blockIdx.x * blockDim.x + threadIdx.x;
    if (i >= NE) return;
    int s = ei[i], d = ei[NE + i];
    float e = a_s[s] + a_d[d];
    e = (e > 0.0f) ? e : 0.2f * e;               // leaky_relu(0.2)
    esc[i] = e;
    atomicMax(&key[d], fkey(e));
}

__global__ void edge_exp(const int* __restrict__ ei, const unsigned* __restrict__ key,
                         float* __restrict__ esc, float* __restrict__ den)
{
    int i = blockIdx.x * blockDim.x + threadIdx.x;
    if (i >= NE) return;
    int d = ei[NE + i];
    float ex = __expf(esc[i] - fkeyinv(key[d]));
    esc[i] = ex;
    atomicAdd(&den[d], ex);
}

__global__ void edge_agg(const int* __restrict__ ei, const float* __restrict__ esc,
                         const float* __restrict__ den, const float* __restrict__ hcat,
                         float* __restrict__ agg)
{
    int gid  = blockIdx.x * blockDim.x + threadIdx.x;
    int lane = gid & 31;
    int w    = gid >> 5;
    int nw   = (gridDim.x * blockDim.x) >> 5;
    for (int i = w; i < NE; i += nw) {
        int s = ei[i], d = ei[NE + i];
        float dn = den[d];
        float al = (dn > 0.0f) ? esc[i] / dn : 0.0f;
        const float* hs = hcat + (size_t)s * 128;       // h_src = cols 0..63
        atomicAdd(&agg[(size_t)d * 64 + lane],      hs[lane]      * al);
        atomicAdd(&agg[(size_t)d * 64 + lane + 32], hs[lane + 32] * al);
    }
}

// ---------------------------------------------------------------------------
// Output GEMM: out = relu(agg + bias_conv) @ W_lin + b_lin   (f16 WMMA)
// ---------------------------------------------------------------------------
__global__ __launch_bounds__(256) void gat_out(
    const float* __restrict__ agg, const float* __restrict__ bias_conv,
    const float* __restrict__ Wlin, const float* __restrict__ blin,
    float* __restrict__ out)
{
    __shared__ _Float16 WlT[64 * 64];   // WlT[c*64 + k] = W_lin[k][c]
    __shared__ float bc[64], bl[64];
    const int tid = threadIdx.x;
    for (int i = tid; i < 64 * 64; i += 256) {
        int c = i & 63, k = i >> 6;
        WlT[c * 64 + k] = (_Float16)Wlin[k * 64 + c];
    }
    if (tid < 64) { bc[tid] = bias_conv[tid]; bl[tid] = blin[tid]; }
    __syncthreads();

    const int wave = tid >> 5;
    const int lane = tid & 31;
    const int h    = lane >> 4;
    const int ln   = lane & 15;
    const int rowbase = blockIdx.x * 128 + wave * 16;

    int arow = rowbase + ln;
    if (arow >= NN) arow = NN - 1;
    const float* ar = agg + (size_t)arow * 64;

    v8f acc[4];
    #pragma unroll
    for (int t = 0; t < 4; ++t)
        #pragma unroll
        for (int r = 0; r < 8; ++r) acc[t][r] = 0.0f;

    #pragma unroll
    for (int kt = 0; kt < 2; ++kt) {
        int base = kt * 32 + 16 * h;
        float4 f0 = *(const float4*)(ar + base);
        float4 f1 = *(const float4*)(ar + base + 4);
        float4 f2 = *(const float4*)(ar + base + 8);
        float4 f3 = *(const float4*)(ar + base + 12);
        v16h a;
        a[0] =(_Float16)fmaxf(f0.x + bc[base + 0],  0.0f);
        a[1] =(_Float16)fmaxf(f0.y + bc[base + 1],  0.0f);
        a[2] =(_Float16)fmaxf(f0.z + bc[base + 2],  0.0f);
        a[3] =(_Float16)fmaxf(f0.w + bc[base + 3],  0.0f);
        a[4] =(_Float16)fmaxf(f1.x + bc[base + 4],  0.0f);
        a[5] =(_Float16)fmaxf(f1.y + bc[base + 5],  0.0f);
        a[6] =(_Float16)fmaxf(f1.z + bc[base + 6],  0.0f);
        a[7] =(_Float16)fmaxf(f1.w + bc[base + 7],  0.0f);
        a[8] =(_Float16)fmaxf(f2.x + bc[base + 8],  0.0f);
        a[9] =(_Float16)fmaxf(f2.y + bc[base + 9],  0.0f);
        a[10]=(_Float16)fmaxf(f2.z + bc[base + 10], 0.0f);
        a[11]=(_Float16)fmaxf(f2.w + bc[base + 11], 0.0f);
        a[12]=(_Float16)fmaxf(f3.x + bc[base + 12], 0.0f);
        a[13]=(_Float16)fmaxf(f3.y + bc[base + 13], 0.0f);
        a[14]=(_Float16)fmaxf(f3.z + bc[base + 14], 0.0f);
        a[15]=(_Float16)fmaxf(f3.w + bc[base + 15], 0.0f);

        #pragma unroll
        for (int nt = 0; nt < 4; ++nt) {
            int n = nt * 16 + ln;
            v16h b = *(const v16h*)(&WlT[n * 64 + kt * 32 + 16 * h]);
            acc[nt] = __builtin_amdgcn_wmma_f32_16x16x32_f16(
                false, a, false, b, (short)0, acc[nt], false, false);
        }
    }

    #pragma unroll
    for (int r = 0; r < 8; ++r) {
        int row = rowbase + 8 * h + r;
        if (row < NN) {
            float* orow = out + (size_t)row * 64;
            #pragma unroll
            for (int nt = 0; nt < 4; ++nt) {
                int col = nt * 16 + ln;
                orow[col] = acc[nt][r] + bl[col];
            }
        }
    }
}

// ---------------------------------------------------------------------------
extern "C" void kernel_launch(void* const* d_in, const int* in_sizes, int n_in,
                              void* d_out, int out_size, void* d_ws, size_t ws_size,
                              hipStream_t stream) {
    (void)in_sizes; (void)n_in; (void)out_size; (void)ws_size;
    const float* x      = (const float*)d_in[0];
    const int*   ei     = (const int*)d_in[1];
    const float* Wsrc   = (const float*)d_in[2];
    const float* Wdst   = (const float*)d_in[3];
    const float* att_s  = (const float*)d_in[4];
    const float* att_d  = (const float*)d_in[5];
    const float* biasc  = (const float*)d_in[6];
    const float* Wlin   = (const float*)d_in[7];
    const float* blin   = (const float*)d_in[8];

    float* ws = (float*)d_ws;
    const size_t N = NN, E = NE;
    float*    hcat = ws;                          // N*128
    float*    agg  = hcat + N * 128;              // N*64  (zeroed)
    unsigned* key  = (unsigned*)(agg + N * 64);   // N     (zeroed: smallest key)
    float*    den  = (float*)(key + N);           // N     (zeroed)
    float*    a_s  = den + N;                     // N
    float*    a_d  = a_s + N;                     // N
    float*    esc  = a_d + N;                     // E (score, then exp in-place)

    size_t nz = N * 66;                           // agg + key + den contiguous
    zero_f<<<(int)((nz + 255) / 256), 256, 0, stream>>>(agg, nz);

    gat_proj<<<(int)((N + 127) / 128), 256, 0, stream>>>(
        x, Wsrc, Wdst, att_s, att_d, hcat, a_s, a_d);

    edge_score<<<(int)((E + 255) / 256), 256, 0, stream>>>(ei, a_s, a_d, esc, key);
    edge_exp  <<<(int)((E + 255) / 256), 256, 0, stream>>>(ei, key, esc, den);
    edge_agg  <<<2048, 256, 0, stream>>>(ei, esc, den, hcat, agg);

    gat_out<<<(int)((N + 127) / 128), 256, 0, stream>>>(
        agg, biasc, Wlin, blin, (float*)d_out);
}
